// MambaBlock_3100966387717
// MI455X (gfx1250) — matrix operand order repaired and checked by reading, not compiled, not run on hardware
//
#include <hip/hip_runtime.h>
#include <hip/hip_bf16.h>
#include <stdint.h>
#include <stddef.h>

#ifndef __has_builtin
#define __has_builtin(x) 0
#endif

#if defined(__HIP_DEVICE_COMPILE__)
#if __has_builtin(__builtin_amdgcn_global_load_async_to_lds_b128) && \
    __has_builtin(__builtin_amdgcn_s_wait_asynccnt)
#define HAVE_ASYNC_LDS 1
#else
#define HAVE_ASYNC_LDS 0
#endif
#else
#define HAVE_ASYNC_LDS 0
#endif

typedef __bf16 bf16;
typedef __attribute__((ext_vector_type(16))) __bf16 v16bf;
typedef __attribute__((ext_vector_type(8)))  __bf16 v8bf;
typedef __attribute__((ext_vector_type(8)))  float   v8f;
typedef int v4i __attribute__((vector_size(16)));   // matches builtin's pointee type

// ---- problem constants ----
namespace {
constexpr int Bq  = 2;
constexpr int Lq  = 1024;
constexpr int DM  = 1024;              // d_model
constexpr int DI  = 2048;              // d_inner
constexpr int DSQ = 16;                // d_state
constexpr int BLq = Bq * Lq;           // 2048 rows for all GEMMs
constexpr int NXZ = 2 * DI;            // 4096
constexpr int NS  = 2 * DSQ + DI;      // 2080 (x_proj cols)
constexpr int NSP = 2176;              // 2080 padded up to a multiple of 128
}

// ---- GEMM tiling ----
#define BMT 128
#define BNT 128
#define BKT 32
#define APITCH 40   // bf16 elems per LDS row (32 + 8 pad, keeps b128 reads conflict-free)
#define BPITCH 40

__device__ __forceinline__ v16bf frag_combine(v8bf lo, v8bf hi) {
  v16bf r;
#pragma unroll
  for (int e = 0; e < 8; ++e) { r[e] = lo[e]; r[e + 8] = hi[e]; }
  return r;
}

__device__ __forceinline__ void async_wait_barrier() {
#if HAVE_ASYNC_LDS
  __builtin_amdgcn_s_wait_asynccnt(0);
#endif
  __syncthreads();
}

// C[M,N](f32) = A[M,K](bf16,row-major,lda) * B[K,N](bf16,row-major,ldb)
// Grid: (N/128, M/128), 256 threads = 8 waves (2x4 wave grid), each wave 64x32.
__global__ __launch_bounds__(256, 2)
void gemm_bf16_wmma_kernel(const bf16* __restrict__ A,
                           const bf16* __restrict__ Bm,
                           float* __restrict__ C,
                           int K, int lda, int ldb, int ldc)
{
  __shared__ __align__(16) bf16 As[2][BMT * APITCH];   // [row][k]
  __shared__ __align__(16) bf16 Bt[2][BNT * BPITCH];   // transposed: [n][k]

  const int t    = threadIdx.x;
  const int lane = t & 31;
  const int wave = t >> 5;
  const int wm   = wave >> 2;    // 0..1
  const int wn   = wave & 3;     // 0..3
  const int half = lane >> 4;    // ISA fragment half (lanes 0-15 / 16-31)
  const int l16  = lane & 15;

  const int m0 = blockIdx.y * BMT;
  const int n0 = blockIdx.x * BNT;

  v8f acc[4][2] = {};
  const int nk = K / BKT;

  auto stage = [&](int buf, int kt) {
    const int k0 = kt * BKT;
    // A tile: 128 rows x 32 bf16 = 512 x 16B chunks, 2 per thread (async DMA to LDS)
#pragma unroll
    for (int j = 0; j < 2; ++j) {
      const int c   = t * 2 + j;
      const int row = c >> 2;
      const int ko  = (c & 3) * 8;
      const bf16* src = A + (size_t)(m0 + row) * lda + (k0 + ko);
      bf16* dst = &As[buf][row * APITCH + ko];
#if HAVE_ASYNC_LDS
      __builtin_amdgcn_global_load_async_to_lds_b128(
          (__attribute__((address_space(1))) v4i*)src,
          (__attribute__((address_space(3))) v4i*)dst, 0, 0);
#else
      *(v8bf*)dst = *(const v8bf*)src;
#endif
    }
    // B tile: 32(K) x 128(N), transposed into LDS so fragments read contiguous K
#pragma unroll
    for (int j = 0; j < 2; ++j) {
      const int c  = t + j * 256;     // 0..511
      const int k  = c & 31;
      const int nc = c >> 5;          // 16 chunks of 8 columns
      const bf16* src = Bm + (size_t)(k0 + k) * ldb + (n0 + nc * 8);
      v8bf v = *(const v8bf*)src;
      bf16* dst = &Bt[buf][(nc * 8) * BPITCH + k];
#pragma unroll
      for (int e = 0; e < 8; ++e) dst[e * BPITCH] = v[e];
    }
  };

  stage(0, 0);
  async_wait_barrier();

  int buf = 0;
  for (int kt = 0; kt < nk; ++kt) {
    if (kt + 1 < nk) stage(buf ^ 1, kt + 1);

    // A fragment (16-bit 16x32): lane l16 = row, half h: K runs h*8..+7 and h*8+16..+23
    v16bf afrag[4];
#pragma unroll
    for (int fm = 0; fm < 4; ++fm) {
      const bf16* ap = &As[buf][(wm * 64 + fm * 16 + l16) * APITCH + half * 8];
      afrag[fm] = frag_combine(*(const v8bf*)ap, *(const v8bf*)(ap + 16));
    }
    // B fragment (32x16): lane l16 = col, half h: K runs h*8..+7 and h*8+16..+23
    v16bf bfrag[2];
#pragma unroll
    for (int fn = 0; fn < 2; ++fn) {
      const bf16* bp = &Bt[buf][(wn * 32 + fn * 16 + l16) * BPITCH + half * 8];
      bfrag[fn] = frag_combine(*(const v8bf*)bp, *(const v8bf*)(bp + 16));
    }
#pragma unroll
    for (int fm = 0; fm < 4; ++fm)
#pragma unroll
      for (int fn = 0; fn < 2; ++fn)
        acc[fm][fn] = __builtin_amdgcn_wmma_f32_16x16x32_bf16(
            false, afrag[fm], false, bfrag[fn], (short)0, acc[fm][fn],
            false, false);

    async_wait_barrier();
    buf ^= 1;
  }

  // C/D layout: VGPR i -> row fm*16 + half*8 + i, col = l16
#pragma unroll
  for (int fm = 0; fm < 4; ++fm) {
#pragma unroll
    for (int fn = 0; fn < 2; ++fn) {
      const int row0 = m0 + wm * 64 + fm * 16 + half * 8;
      const int col  = n0 + wn * 32 + fn * 16 + l16;
      float* cp = C + (size_t)row0 * ldc + col;
#pragma unroll
      for (int i = 0; i < 8; ++i)
        cp[(size_t)i * ldc] = acc[fm][fn][i];
    }
  }
}

// ---- elementwise / conversion kernels ----
__global__ void f32_to_bf16_kernel(const float* __restrict__ s, bf16* __restrict__ d, int n) {
  int i = blockIdx.x * blockDim.x + threadIdx.x;
  if (i < n) d[i] = (bf16)s[i];
}

__global__ void f32_to_bf16_pad_kernel(const float* __restrict__ s, bf16* __restrict__ d,
                                       int rows, int scols, int dcols) {
  int i = blockIdx.x * blockDim.x + threadIdx.x;
  if (i >= rows * dcols) return;
  int r = i / dcols, c = i % dcols;
  d[i] = (c < scols) ? (bf16)s[(size_t)r * scols + c] : (bf16)0.0f;
}

// depthwise causal conv (k=4, cross-correlation, left pad 3) + bias + SiLU
__global__ void conv_silu_kernel(const float* __restrict__ xz,
                                 const float* __restrict__ cw,
                                 const float* __restrict__ cb,
                                 float* __restrict__ u,
                                 bf16* __restrict__ u16) {
  int i = blockIdx.x * blockDim.x + threadIdx.x;
  if (i >= BLq * DI) return;
  int d  = i % DI;
  int bl = i / DI;
  int l  = bl % Lq;
  float acc = cb[d];
#pragma unroll
  for (int j = 0; j < 4; ++j) {
    int ls = l - 3 + j;
    if (ls >= 0)
      acc = fmaf(cw[d * 4 + j], xz[(size_t)(bl - 3 + j) * NXZ + d], acc);
  }
  float sv = acc / (1.0f + __expf(-acc));
  u[i]   = sv;
  u16[i] = (bf16)sv;
}

__global__ void dt_softplus_kernel(float* __restrict__ dt, const float* __restrict__ bias) {
  int i = blockIdx.x * blockDim.x + threadIdx.x;
  if (i >= BLq * DI) return;
  float v = dt[i] + bias[i % DI];
  dt[i] = (v > 20.0f) ? v : log1pf(__expf(v));
}

// selective scan: one thread per (b,d) channel, 16 states in registers,
// epilogue (y + D*u) * silu(z) fused, output bf16 for the final GEMM.
__global__ void ssm_scan_kernel(const float* __restrict__ dt,
                                const float* __restrict__ u,
                                const float* __restrict__ ssm,   // padded, cols 0..15=B, 16..31=C
                                const float* __restrict__ xz,    // z at cols DI..2*DI-1
                                const float* __restrict__ A_log,
                                const float* __restrict__ Dp,
                                bf16* __restrict__ y16) {
  int tid = blockIdx.x * blockDim.x + threadIdx.x;
  if (tid >= Bq * DI) return;
  int b = tid / DI;
  int d = tid % DI;
  float Ar[DSQ], h[DSQ];
#pragma unroll
  for (int n = 0; n < DSQ; ++n) { Ar[n] = -__expf(A_log[d * DSQ + n]); h[n] = 0.0f; }
  float Dd = Dp[d];
  for (int l = 0; l < Lq; ++l) {
    int bl = b * Lq + l;
    float dtv = dt[(size_t)bl * DI + d];
    float uv  = u[(size_t)bl * DI + d];
    float zv  = xz[(size_t)bl * NXZ + DI + d];
    const float* Brow = ssm + (size_t)bl * NSP;
    const float* Crow = Brow + DSQ;
    float y = 0.0f;
#pragma unroll
    for (int n = 0; n < DSQ; ++n) {
      float dA = __expf(dtv * Ar[n]);
      h[n] = fmaf(dA, h[n], dtv * Brow[n] * uv);
      y = fmaf(h[n], Crow[n], y);
    }
    y = fmaf(Dd, uv, y);
    y *= zv / (1.0f + __expf(-zv));     // gate with silu(z)
    y16[(size_t)bl * DI + d] = (bf16)y;
  }
}

extern "C" void kernel_launch(void* const* d_in, const int* in_sizes, int n_in,
                              void* d_out, int out_size, void* d_ws, size_t ws_size,
                              hipStream_t stream) {
  const float* x        = (const float*)d_in[0];
  const float* in_w     = (const float*)d_in[1];
  const float* conv_w   = (const float*)d_in[2];
  const float* conv_b   = (const float*)d_in[3];
  const float* xproj_w  = (const float*)d_in[4];
  const float* dtproj_w = (const float*)d_in[5];
  const float* dtproj_b = (const float*)d_in[6];
  const float* A_log    = (const float*)d_in[7];
  const float* Dp       = (const float*)d_in[8];
  const float* out_w    = (const float*)d_in[9];
  float* out = (float*)d_out;
  (void)in_sizes; (void)n_in; (void)out_size; (void)ws_size;

  char* ws = (char*)d_ws;
  size_t off = 0;
  auto wsalloc = [&](size_t bytes) {
    void* p = ws + off;
    off = (off + bytes + 255) & ~(size_t)255;
    return p;
  };
  bf16*  x16    = (bf16*) wsalloc((size_t)BLq * DM  * 2);
  bf16*  winp16 = (bf16*) wsalloc((size_t)DM  * NXZ * 2);
  bf16*  wxp16  = (bf16*) wsalloc((size_t)DI  * NSP * 2);
  bf16*  wdt16  = (bf16*) wsalloc((size_t)DI  * DI  * 2);
  bf16*  wout16 = (bf16*) wsalloc((size_t)DI  * DM  * 2);
  float* xz     = (float*)wsalloc((size_t)BLq * NXZ * 4);
  float* uf     = (float*)wsalloc((size_t)BLq * DI  * 4);
  bf16*  u16    = (bf16*) wsalloc((size_t)BLq * DI  * 2);
  float* ssm    = (float*)wsalloc((size_t)BLq * NSP * 4);
  bf16*  ssm16  = (bf16*) wsalloc((size_t)BLq * NSP * 2);
  float* dtb    = (float*)wsalloc((size_t)BLq * DI  * 4);
  bf16*  y16    = (bf16*) wsalloc((size_t)BLq * DI  * 2);

  const int TPB = 256;
  auto nb = [](size_t n) { return (unsigned)((n + 255) / 256); };

  // f32 -> bf16 conversions (x_proj zero-padded 2080 -> 2176 cols)
  f32_to_bf16_kernel<<<nb((size_t)BLq * DM),  TPB, 0, stream>>>(x, x16, BLq * DM);
  f32_to_bf16_kernel<<<nb((size_t)DM * NXZ),  TPB, 0, stream>>>(in_w, winp16, DM * NXZ);
  f32_to_bf16_pad_kernel<<<nb((size_t)DI * NSP), TPB, 0, stream>>>(xproj_w, wxp16, DI, NS, NSP);
  f32_to_bf16_kernel<<<nb((size_t)DI * DI),   TPB, 0, stream>>>(dtproj_w, wdt16, DI * DI);
  f32_to_bf16_kernel<<<nb((size_t)DI * DM),   TPB, 0, stream>>>(out_w, wout16, DI * DM);

  // xz = x @ in_proj_w                     [2048,1024]x[1024,4096]
  gemm_bf16_wmma_kernel<<<dim3(NXZ / BNT, BLq / BMT), TPB, 0, stream>>>(
      x16, winp16, xz, DM, DM, NXZ, NXZ);

  // u = silu(depthwise_conv(x_inner) + b)
  conv_silu_kernel<<<nb((size_t)BLq * DI), TPB, 0, stream>>>(xz, conv_w, conv_b, uf, u16);

  // ssm = u @ x_proj_w (padded)            [2048,2048]x[2048,2176]
  gemm_bf16_wmma_kernel<<<dim3(NSP / BNT, BLq / BMT), TPB, 0, stream>>>(
      u16, wxp16, ssm, DI, DI, NSP, NSP);

  f32_to_bf16_kernel<<<nb((size_t)BLq * NSP), TPB, 0, stream>>>(ssm, ssm16, BLq * NSP);

  // dt_raw = ssm[:,32:2080] @ dt_proj_w    [2048,2048]x[2048,2048]
  gemm_bf16_wmma_kernel<<<dim3(DI / BNT, BLq / BMT), TPB, 0, stream>>>(
      ssm16 + 2 * DSQ, wdt16, dtb, DI, NSP, DI, DI);

  dt_softplus_kernel<<<nb((size_t)BLq * DI), TPB, 0, stream>>>(dtb, dtproj_b);

  // sequential selective scan + gating, outputs bf16 y
  ssm_scan_kernel<<<nb((size_t)Bq * DI), TPB, 0, stream>>>(
      dtb, uf, ssm, xz, A_log, Dp, y16);

  // out = y @ out_proj_w                   [2048,2048]x[2048,1024]
  gemm_bf16_wmma_kernel<<<dim3(DM / BNT, BLq / BMT), TPB, 0, stream>>>(
      y16, wout16, out, DI, DI, DM, DM);
}
